// LSTM_292057776562
// MI455X (gfx1250) — matrix operand order repaired and checked by reading, not compiled, not run on hardware
//
#include <hip/hip_runtime.h>

// ---------------- types for CDNA5 WMMA ----------------
typedef __attribute__((ext_vector_type(16))) __bf16 v16bf;
typedef __attribute__((ext_vector_type(8)))  __bf16 bf16x8;
typedef __attribute__((ext_vector_type(8)))  float  v8f;

// ---------------- problem constants ----------------
constexpr int Bn  = 64;          // batch
constexpr int Tn  = 256;         // timesteps
constexpr int Hn  = 64;          // hidden
constexpr int Gn  = 4 * Hn;      // 4H = 256 gate columns
constexpr int INn = 256;         // input size
constexpr int Vn  = 8000;        // vocab
constexpr int Mn  = Bn * Tn;     // 16384 rows for the batched GEMMs

// ---------------- helpers ----------------
__device__ inline unsigned short f2bf(float f) {
  unsigned int u = __float_as_uint(f);
  return (unsigned short)((u + 0x7FFFu + ((u >> 16) & 1u)) >> 16); // RNE
}
__device__ inline float fsigmoid(float x) { return 1.0f / (1.0f + __expf(-x)); }
__device__ inline float ftanh(float x)    { float e = __expf(2.0f * x); return (e - 1.0f) / (e + 1.0f); }

// Build a v16bf fragment from two 16-byte chunks (p0 -> elems 0..7, p1 -> elems 8..15)
__device__ inline v16bf ld16frag(const unsigned short* p0, const unsigned short* p1) {
  bf16x8 lo = *(const bf16x8*)p0;
  bf16x8 hi = *(const bf16x8*)p1;
  v16bf a;
#pragma unroll
  for (int i = 0; i < 8; i++) { a[i] = lo[i]; a[i + 8] = hi[i]; }
  return a;
}

// ---------------- f32 -> bf16 conversion ----------------
__global__ void cvt_f32_to_bf16(const float* __restrict__ in,
                                unsigned short* __restrict__ out, int n) {
  int i = blockIdx.x * blockDim.x + threadIdx.x;
  if (i < n) out[i] = f2bf(in[i]);
}

// ---------------- WMMA GEMM:  C[M,N] = A[M,K] * W[N,K]^T + bias0 + bias1 ----
// One wave (32 threads) per 16x64 output strip (4 N-tiles sharing each A fragment).
// grid = (N/64, M/16). Output written with non-temporal stores (streamed, > L2).
__global__ __launch_bounds__(32)
void gemm_bf16_wmma(const unsigned short* __restrict__ A, int lda,
                    const unsigned short* __restrict__ W, int ldw,
                    const float* __restrict__ bias0,
                    const float* __restrict__ bias1,
                    float* __restrict__ C, int ldc, int K) {
  const int lane = threadIdx.x;
  const int half = lane >> 4;     // 0: lanes 0-15, 1: lanes 16-31
  const int nl   = lane & 15;
  const int n0   = blockIdx.x * 64;
  const int m0   = blockIdx.y * 16;

  const unsigned short* arow = A + (size_t)(m0 + nl) * lda;
  const unsigned short* wrow = W + (size_t)(n0 + nl) * ldw;
  const size_t wstep = (size_t)16 * ldw;  // 16 N-rows between tiles

  v8f acc[4] = {};
  for (int k = 0; k < K; k += 32) {
    // A fragment (16x32 bf16): lane holds row (lane&15); K chunks per ISA layout
    const unsigned short* ap = arow + k + half * 8;
    v16bf a = ld16frag(ap, ap + 16);
#pragma unroll
    for (int j = 0; j < 4; j++) {
      // B fragment (32x16 bf16): lane holds col (lane&15); K chunks per ISA layout
      const unsigned short* bp = wrow + (size_t)j * wstep + k + half * 8;
      v16bf b = ld16frag(bp, bp + 16);
      acc[j] = __builtin_amdgcn_wmma_f32_16x16x32_bf16(false, a, false, b,
                                                       (short)0, acc[j], false, false);
    }
  }

#pragma unroll
  for (int j = 0; j < 4; j++) {
    const int n = n0 + j * 16 + nl;
    float bv = 0.0f;
    if (bias0) bv += bias0[n];
    if (bias1) bv += bias1[n];
#pragma unroll
    for (int r = 0; r < 8; r++) {
      // C layout: VGPR r -> row m0 + r + 8*half, col n
      __builtin_nontemporal_store(acc[j][r] + bv,
                                  &C[(size_t)(m0 + r + half * 8) * ldc + n]);
    }
  }
}

// ---------------- LSTM recurrence ----------------
// One wave per block; block b handles batch rows [16b, 16b+16).
// gx  : precomputed x-projection + biases, f32 [B*T, 256], row index = batch*T + t
// Whh : bf16 [256][64]  (gate-major, K contiguous -> B^T fragments are contiguous)
// hout: bf16 [B*T, 64]
__global__ __launch_bounds__(32)
void lstm_recurrence(const float* __restrict__ gx,
                     const unsigned short* __restrict__ Whh,
                     unsigned short* __restrict__ hout) {
  __shared__ __align__(16) unsigned short whs[Gn * Hn];   // 32 KB
  __shared__ __align__(16) unsigned short hbuf[16 * Hn];  //  2 KB
  __shared__ __align__(16) float          gbuf[16 * Gn];  // 16 KB

  const int lane = threadIdx.x;
  const int m0   = blockIdx.x * 16;
  const int half = lane >> 4;
  const int nl   = lane & 15;

  // Stage W_hh into LDS (resident for all 256 steps)
  {
    const uint4* src = (const uint4*)Whh;
    uint4* dst = (uint4*)whs;
    for (int i = lane; i < (Gn * Hn) / 8; i += 32) dst[i] = src[i];
  }
  for (int i = lane; i < 16 * Hn; i += 32) hbuf[i] = 0;
  __syncthreads();

  // Cell state in registers: lane covers (m = lane>>1, j = (lane&1)*32 + q)
  float c[32];
#pragma unroll
  for (int q = 0; q < 32; q++) c[q] = 0.0f;
  const int am  = lane >> 1;
  const int aj0 = (lane & 1) * 32;

  for (int t = 0; t < Tn; t++) {
    // Prefetch next timestep's 16KB gate slab (hides HBM/L2 latency behind
    // this step's WMMA + activation chain). 16 rows x 1KB; 4 lines per lane.
    if (t + 1 < Tn) {
      const char* pbase =
          (const char*)(gx + (((size_t)(m0 + nl) * Tn) + (t + 1)) * Gn) + half * 512;
#pragma unroll
      for (int p = 0; p < 4; p++) __builtin_prefetch(pbase + p * 128, 0, 1);
    }

    // ---- phase 1: gates[16,256] = xproj + h_{t-1} @ Whh^T via WMMA ----
#pragma unroll 1
    for (int nt = 0; nt < 16; nt++) {
      v8f acc;
#pragma unroll
      for (int r = 0; r < 8; r++)
        acc[r] = gx[(((size_t)(m0 + r + half * 8) * Tn + t) * Gn) + nt * 16 + nl];
#pragma unroll
      for (int kb = 0; kb < 2; kb++) {
        const unsigned short* hp = &hbuf[nl * Hn + kb * 32 + half * 8];
        v16bf a = ld16frag(hp, hp + 16);
        const unsigned short* bp = &whs[(size_t)(nt * 16 + nl) * Hn + kb * 32 + half * 8];
        v16bf b = ld16frag(bp, bp + 16);
        acc = __builtin_amdgcn_wmma_f32_16x16x32_bf16(false, a, false, b,
                                                      (short)0, acc, false, false);
      }
#pragma unroll
      for (int r = 0; r < 8; r++)
        gbuf[(r + half * 8) * Gn + nt * 16 + nl] = acc[r];
    }
    __syncthreads();

    // ---- phase 2: activations + state update (f32 VALU/transcendentals) ----
#pragma unroll
    for (int q = 0; q < 32; q++) {
      const int j = aj0 + q;
      float gi = gbuf[am * Gn + j];
      float gf = gbuf[am * Gn + 64 + j];
      float gg = gbuf[am * Gn + 128 + j];
      float go = gbuf[am * Gn + 192 + j];
      float ii = fsigmoid(gi);
      float ff = fsigmoid(gf);
      float tg = ftanh(gg);
      float oo = fsigmoid(go);
      float cn = ff * c[q] + ii * tg;
      c[q] = cn;
      unsigned short hb = f2bf(oo * ftanh(cn));
      hbuf[am * Hn + j] = hb;
      hout[((size_t)(m0 + am) * Tn + t) * Hn + j] = hb;
    }
    __syncthreads();
  }
}

// ---------------- host-side orchestration ----------------
extern "C" void kernel_launch(void* const* d_in, const int* in_sizes, int n_in,
                              void* d_out, int out_size, void* d_ws, size_t ws_size,
                              hipStream_t stream) {
  const float* x    = (const float*)d_in[0];
  const float* Wih0 = (const float*)d_in[1];
  const float* Whh0 = (const float*)d_in[2];
  const float* bih0 = (const float*)d_in[3];
  const float* bhh0 = (const float*)d_in[4];
  const float* Wih1 = (const float*)d_in[5];
  const float* Whh1 = (const float*)d_in[6];
  const float* bih1 = (const float*)d_in[7];
  const float* bhh1 = (const float*)d_in[8];
  const float* Wl   = (const float*)d_in[9];
  const float* bl   = (const float*)d_in[10];
  float* out = (float*)d_out;

  char* ws = (char*)d_ws;
  size_t off = 0;
  auto alloc = [&](size_t bytes) -> void* {
    void* p = ws + off;
    off = (off + bytes + 255) & ~(size_t)255;
    return p;
  };

  unsigned short* xb    = (unsigned short*)alloc((size_t)Mn * INn * 2); // 8 MB
  unsigned short* wih0b = (unsigned short*)alloc((size_t)Gn * INn * 2);
  unsigned short* whh0b = (unsigned short*)alloc((size_t)Gn * Hn * 2);
  unsigned short* wih1b = (unsigned short*)alloc((size_t)Gn * Hn * 2);
  unsigned short* whh1b = (unsigned short*)alloc((size_t)Gn * Hn * 2);
  unsigned short* wlb   = (unsigned short*)alloc((size_t)Vn * Hn * 2); // 1 MB
  float*          gx0   = (float*)alloc((size_t)Mn * Gn * 4);          // 16 MB
  unsigned short* h1b   = (unsigned short*)alloc((size_t)Mn * Hn * 2); // 2 MB
  float*          gx1   = (float*)alloc((size_t)Mn * Gn * 4);          // 16 MB
  unsigned short* h2b   = (unsigned short*)alloc((size_t)Mn * Hn * 2); // 2 MB

  auto cvt = [&](const float* src, unsigned short* dst, int n) {
    cvt_f32_to_bf16<<<(n + 255) / 256, 256, 0, stream>>>(src, dst, n);
  };
  cvt(x,    xb,    Mn * INn);
  cvt(Wih0, wih0b, Gn * INn);
  cvt(Whh0, whh0b, Gn * Hn);
  cvt(Wih1, wih1b, Gn * Hn);
  cvt(Whh1, whh1b, Gn * Hn);
  cvt(Wl,   wlb,   Vn * Hn);

  // Layer 0 input projection: gx0 = x @ Wih0^T + bih0 + bhh0   [16384 x 256]
  gemm_bf16_wmma<<<dim3(Gn / 64, Mn / 16), 32, 0, stream>>>(
      xb, INn, wih0b, INn, bih0, bhh0, gx0, Gn, INn);

  // Layer 0 recurrence -> h1 (bf16)
  lstm_recurrence<<<dim3(Bn / 16), 32, 0, stream>>>(gx0, whh0b, h1b);

  // Layer 1 input projection: gx1 = h1 @ Wih1^T + bih1 + bhh1
  gemm_bf16_wmma<<<dim3(Gn / 64, Mn / 16), 32, 0, stream>>>(
      h1b, Hn, wih1b, Hn, bih1, bhh1, gx1, Gn, Hn);

  // Layer 1 recurrence -> h2 (bf16)
  lstm_recurrence<<<dim3(Bn / 16), 32, 0, stream>>>(gx1, whh1b, h2b);

  // Logits: out = h2 @ Wl^T + bl   [16384 x 8000]  (store-bandwidth bound)
  gemm_bf16_wmma<<<dim3(Vn / 64, Mn / 16), 32, 0, stream>>>(
      h2b, Hn, wlb, Hn, bl, nullptr, out, Vn, Hn);
}
